// CrossAttention_57131654971854
// MI455X (gfx1250) — matrix operand order repaired and checked by reading, not compiled, not run on hardware
//
#include <hip/hip_runtime.h>

typedef __bf16 bf16_t;
typedef __attribute__((ext_vector_type(16))) __bf16 v16bf;
typedef __attribute__((ext_vector_type(8)))  __bf16 v8bf;
typedef __attribute__((ext_vector_type(8)))  float  v8f;
typedef __attribute__((ext_vector_type(4)))  unsigned int u32x4;
typedef __attribute__((ext_vector_type(8)))  unsigned int u32x8;

#define BDIM   4
#define NSEQ   2048
#define DMODEL 1024
#define NHEADS 16
#define DH     64
#define ATT_SCALE 0.125f   // 64^-0.5

__device__ __forceinline__ v8f wmma_bf16(v16bf a, v16bf b, v8f c) {
  return __builtin_amdgcn_wmma_f32_16x16x32_bf16(false, a, false, b, (short)0, c,
                                                 false, false);
}

__device__ __forceinline__ v16bf cat16(v8bf lo, v8bf hi) {
  return __builtin_shufflevector(lo, hi, 0, 1, 2, 3, 4, 5, 6, 7,
                                 8, 9, 10, 11, 12, 13, 14, 15);
}

// A-matrix 16x32 bf16 fragment from row-major source (leading dim ld, elements).
// lanes 0-15: row=lane, K=0..7 & 16..23 ; lanes 16-31: row=lane-16, K=8..15 & 24..31.
__device__ __forceinline__ v16bf load_a_frag(const bf16_t* p, int ld) {
  const int lane = threadIdx.x & 31;
  const bf16_t* r = p + (lane & 15) * ld + (lane >> 4) * 8;
  return cat16(*(const v8bf*)r, *(const v8bf*)(r + 16));
}

// B-matrix 32x16 fragment from a K-contiguous source: B(k,n) = p[n*ld + k].
// lanes 0-15: col=lane, K=0..7 & 16..23 ; lanes 16-31: col=lane-16, K=8..15 & 24..31.
__device__ __forceinline__ v16bf load_b_frag_t(const bf16_t* p, int ld) {
  const int lane = threadIdx.x & 31;
  const bf16_t* r = p + (lane & 15) * ld + (lane >> 4) * 8;
  return cat16(*(const v8bf*)r, *(const v8bf*)(r + 16));
}

// ---- Tensor Data Mover: 2D bf16 tile (tile_x contiguous elems * tile_y rows) ----
__device__ __forceinline__ void tdm_load_2d_bf16(unsigned lds_off, const void* gptr,
                                                 unsigned tile_x, unsigned tile_y,
                                                 unsigned tensor_x, unsigned tensor_y,
                                                 unsigned stride_elems) {
  unsigned long long ga = (unsigned long long)gptr;
  u32x4 g0;
  g0[0] = 1u;                                      // count=1 (valid user D#)
  g0[1] = lds_off;                                 // lds_addr
  g0[2] = (unsigned)(ga & 0xFFFFFFFFu);            // global_addr[31:0]
  g0[3] = (unsigned)((ga >> 32) & 0x01FFFFFFu)     // global_addr[56:32]
        | 0x80000000u;                             // type=2 ("image")
  u32x8 g1;
  g1[0] = 1u << 16;                                // wg_mask=0, data_size=1 (2B)
  g1[1] = (tensor_x & 0xFFFFu) << 16;              // tensor_dim0[15:0]
  g1[2] = ((tensor_x >> 16) & 0xFFFFu) | ((tensor_y & 0xFFFFu) << 16);
  g1[3] = ((tensor_y >> 16) & 0xFFFFu) | (tile_x << 16);   // tile_dim0
  g1[4] = tile_y & 0xFFFFu;                        // tile_dim1 (tile_dim2=0)
  g1[5] = stride_elems;                            // tensor_dim0_stride[31:0]
  g1[6] = 0;
  g1[7] = 0;
  asm volatile("tensor_load_to_lds %0, %1" :: "s"(g0), "s"(g1) : "memory");
}

__global__ void f32_to_bf16_kernel(const float* __restrict__ in,
                                   bf16_t* __restrict__ out, int n) {
  int i = blockIdx.x * blockDim.x + threadIdx.x;
  if (i < n) out[i] = (bf16_t)in[i];
}

// 1024x1024 transpose + convert: out[c][r] = in[r][c] (weights become [out_col][k]).
__global__ void f32_to_bf16_T_kernel(const float* __restrict__ in,
                                     bf16_t* __restrict__ out) {
  int i = blockIdx.x * blockDim.x + threadIdx.x;
  int r = i >> 10, c = i & 1023;
  out[(c << 10) + r] = (bf16_t)in[i];
}

// ---------------- GEMM main-loop building blocks (double-buffered) ----------------
// Block = 256 threads = 8 waves; tile 128(M) x 64(N); K step 32.
// A tile (128x32, row-major image) arrives via TDM; B tile (64 rows of pre-transposed
// weights, [n][k]) via coalesced b128 global->LDS stores. Ping-pong on two buffers so
// the DMA and global loads for tile t+1 overlap the WMMAs on tile t.

#define GEMM_BODY(X_, WT_, M0_, N0_, MTOT_)                                          \
  const int wave = threadIdx.x >> 5;                                                 \
  const int lane = threadIdx.x & 31;                                                 \
  const int bn = threadIdx.x >> 2, bc = (threadIdx.x & 3) * 8;                       \
  v8f zero = {};                                                                     \
  v8f acc[4] = {zero, zero, zero, zero};                                             \
  /* prologue: stage tile k0=0 into buffer 0 */                                      \
  *(uint4*)(&BsT[0][bn][bc]) = *(const uint4*)(&WT_[(size_t)(N0_ + bn) * DMODEL + bc]); \
  if (wave == 0) {                                                                   \
    tdm_load_2d_bf16((unsigned)(size_t)(&As[0][0][0]),                               \
                     X_ + (size_t)M0_ * DMODEL, 32, 128, DMODEL, MTOT_, DMODEL);     \
    __builtin_amdgcn_s_wait_tensorcnt(0);                                            \
  }                                                                                  \
  __syncthreads();                                                                   \
  int buf = 0;                                                                       \
  for (int k0 = 0; k0 < DMODEL; k0 += 32) {                                          \
    const int nbuf = buf ^ 1;                                                        \
    if (k0 + 32 < DMODEL) {                                                          \
      *(uint4*)(&BsT[nbuf][bn][bc]) =                                                \
          *(const uint4*)(&WT_[(size_t)(N0_ + bn) * DMODEL + k0 + 32 + bc]);         \
      if (wave == 0)                                                                 \
        tdm_load_2d_bf16((unsigned)(size_t)(&As[nbuf][0][0]),                        \
                         X_ + (size_t)M0_ * DMODEL + k0 + 32,                        \
                         32, 128, DMODEL, MTOT_, DMODEL);                            \
      __builtin_prefetch(&WT_[(size_t)(N0_ + bn) * DMODEL + k0 + 64 + bc], 0, 3);    \
    }                                                                                \
    v16bf a = load_a_frag(&As[buf][wave * 16][0], 32);                               \
    _Pragma("unroll")                                                                \
    for (int f = 0; f < 4; ++f) {                                                    \
      v16bf b = load_b_frag_t(&BsT[buf][f * 16][0], 32);                             \
      acc[f] = wmma_bf16(a, b, acc[f]);                                              \
    }                                                                                \
    if (wave == 0) __builtin_amdgcn_s_wait_tensorcnt(0);                             \
    __syncthreads();                                                                 \
    buf = nbuf;                                                                      \
  }

// Fused QKV projection. X:[8192,1024] bf16, WT pre-transposed [out][k] bf16.
// z=0 -> Q [B,H,N,64], z=1 -> K [B,H,N,64], z=2 -> V transposed [B,H,64,N].
__global__ __launch_bounds__(256) void qkv_gemm_kernel(
    const bf16_t* __restrict__ X,
    const bf16_t* __restrict__ WqT, const bf16_t* __restrict__ WkT,
    const bf16_t* __restrict__ WvT,
    bf16_t* __restrict__ Q, bf16_t* __restrict__ K, bf16_t* __restrict__ Vt) {
  __shared__ __align__(16) bf16_t As [2][128][32];
  __shared__ __align__(16) bf16_t BsT[2][64][32];

  const int n0 = blockIdx.x * 64;
  const int m0 = blockIdx.y * 128;
  const int z  = blockIdx.z;
  const bf16_t* WT = (z == 0) ? WqT : (z == 1) ? WkT : WvT;

  GEMM_BODY(X, WT, m0, n0, BDIM * NSEQ)

  const int cl = lane & 15, rh = (lane >> 4) * 8;
#pragma unroll
  for (int f = 0; f < 4; ++f) {
#pragma unroll
    for (int j = 0; j < 8; ++j) {
      int m   = m0 + wave * 16 + rh + j;
      int col = n0 + f * 16 + cl;
      int b   = m >> 11, n = m & (NSEQ - 1);
      int h   = col >> 6, d = col & (DH - 1);
      size_t bh = (size_t)((b << 4) | h);
      if (z == 0)
        Q[(bh * NSEQ + n) * DH + d] = (bf16_t)acc[f][j];
      else if (z == 1)
        K[(bh * NSEQ + n) * DH + d] = (bf16_t)acc[f][j];
      else
        Vt[(bh * DH + d) * NSEQ + n] = (bf16_t)acc[f][j];   // V stored [B,H,64,N]
    }
  }
}

// Output projection: Y = AO[8192,1024] @ Wo + bo (WoT pre-transposed), fp32 result.
__global__ __launch_bounds__(256) void out_gemm_kernel(
    const bf16_t* __restrict__ X, const bf16_t* __restrict__ WT,
    const float* __restrict__ bias, float* __restrict__ Y) {
  __shared__ __align__(16) bf16_t As [2][128][32];
  __shared__ __align__(16) bf16_t BsT[2][64][32];

  const int n0 = blockIdx.x * 64;
  const int m0 = blockIdx.y * 128;

  GEMM_BODY(X, WT, m0, n0, BDIM * NSEQ)

  const int cl = lane & 15, rh = (lane >> 4) * 8;
#pragma unroll
  for (int f = 0; f < 4; ++f) {
    int col = n0 + f * 16 + cl;
    float bv = bias[col];
#pragma unroll
    for (int j = 0; j < 8; ++j) {
      int m = m0 + wave * 16 + rh + j;
      Y[(size_t)m * DMODEL + col] = acc[f][j] + bv;
    }
  }
}

// Flash attention: grid (N/64, B*H); 4 waves, each owns a 16-row Q strip.
// K is [B,H,N,64] (K^T fragments contiguous per lane); V is [B,H,64,N] (contiguous).
__global__ __launch_bounds__(128) void attn_kernel(
    const bf16_t* __restrict__ Q, const bf16_t* __restrict__ K,
    const bf16_t* __restrict__ Vt, bf16_t* __restrict__ AO) {
  __shared__ __align__(16) bf16_t Ps[4][16][64];  // per-wave P tile (C-layout -> A-layout)

  const int wave = threadIdx.x >> 5;
  const int lane = threadIdx.x & 31;
  const int cl = lane & 15;
  const int rh = (lane >> 4) * 8;

  const int bh   = blockIdx.y;
  const int bidx = bh >> 4;
  const int h    = bh & 15;
  const bf16_t* Qh  = Q  + (size_t)bh * NSEQ * DH;
  const bf16_t* Kh  = K  + (size_t)bh * NSEQ * DH;
  const bf16_t* Vth = Vt + (size_t)bh * DH * NSEQ;

  const int q0 = blockIdx.x * 64 + wave * 16;

  v16bf qa[2];
#pragma unroll
  for (int kc = 0; kc < 2; ++kc) {
    qa[kc] = load_a_frag(&Qh[(size_t)q0 * DH + kc * 32], DH);
#pragma unroll
    for (int i = 0; i < 16; ++i)
      qa[kc][i] = (bf16_t)((float)qa[kc][i] * ATT_SCALE);
  }

  v8f zero = {};
  v8f o[4] = {zero, zero, zero, zero};
  float m_run[8], l_run[8];
#pragma unroll
  for (int j = 0; j < 8; ++j) { m_run[j] = -1e30f; l_run[j] = 0.f; }

  for (int t = 0; t < NSEQ / 64; ++t) {
    const int kv = t * 64;
    if (t + 1 < NSEQ / 64) {  // near-scope prefetch of next K tile
      __builtin_prefetch(&Kh[(size_t)(kv + 64 + lane) * DH], 0, 3);
      __builtin_prefetch(&Vth[(size_t)lane * NSEQ + kv + 64], 0, 3);
    }
    v8f s[4] = {zero, zero, zero, zero};
#pragma unroll
    for (int f = 0; f < 4; ++f) {
#pragma unroll
      for (int kc = 0; kc < 2; ++kc) {
        v16bf bk = load_b_frag_t(&Kh[(size_t)(kv + f * 16) * DH + kc * 32], DH);
        s[f] = wmma_bf16(qa[kc], bk, s[f]);
      }
    }
    float mnew[8], alpha[8];
#pragma unroll
    for (int j = 0; j < 8; ++j) {
      float mx = fmaxf(fmaxf(s[0][j], s[1][j]), fmaxf(s[2][j], s[3][j]));
      mx = fmaxf(mx, __shfl_xor(mx, 1));
      mx = fmaxf(mx, __shfl_xor(mx, 2));
      mx = fmaxf(mx, __shfl_xor(mx, 4));
      mx = fmaxf(mx, __shfl_xor(mx, 8));
      mnew[j]  = fmaxf(m_run[j], mx);
      alpha[j] = __expf(m_run[j] - mnew[j]);
      m_run[j] = mnew[j];
    }
#pragma unroll
    for (int j = 0; j < 8; ++j) {
      float rs = 0.f;
#pragma unroll
      for (int f = 0; f < 4; ++f) {
        float p = __expf(s[f][j] - mnew[j]);
        s[f][j] = p;
        rs += p;
      }
      rs += __shfl_xor(rs, 1);
      rs += __shfl_xor(rs, 2);
      rs += __shfl_xor(rs, 4);
      rs += __shfl_xor(rs, 8);
      l_run[j] = l_run[j] * alpha[j] + rs;
#pragma unroll
      for (int f = 0; f < 4; ++f) o[f][j] *= alpha[j];
    }
#pragma unroll
    for (int f = 0; f < 4; ++f)
#pragma unroll
      for (int j = 0; j < 8; ++j)
        Ps[wave][rh + j][f * 16 + cl] = (bf16_t)s[f][j];
    __syncthreads();
#pragma unroll
    for (int kc = 0; kc < 2; ++kc) {
      v16bf pa = load_a_frag(&Ps[wave][0][kc * 32], 64);
#pragma unroll
      for (int f = 0; f < 4; ++f) {
        v16bf bv = load_b_frag_t(&Vth[(size_t)(f * 16) * NSEQ + kv + kc * 32], NSEQ);
        o[f] = wmma_bf16(pa, bv, o[f]);
      }
    }
    __syncthreads();
  }

#pragma unroll
  for (int j = 0; j < 8; ++j) {
    float inv = 1.0f / l_run[j];
#pragma unroll
    for (int f = 0; f < 4; ++f) o[f][j] *= inv;
  }
#pragma unroll
  for (int f = 0; f < 4; ++f)
#pragma unroll
    for (int j = 0; j < 8; ++j) {
      int n = q0 + rh + j;
      AO[((size_t)bidx * NSEQ + n) * DMODEL + h * DH + f * 16 + cl] = (bf16_t)o[f][j];
    }
}

extern "C" void kernel_launch(void* const* d_in, const int* in_sizes, int n_in,
                              void* d_out, int out_size, void* d_ws, size_t ws_size,
                              hipStream_t stream) {
  (void)in_sizes; (void)n_in; (void)out_size; (void)ws_size;
  const float* x  = (const float*)d_in[0];
  const float* Wq = (const float*)d_in[1];
  const float* Wk = (const float*)d_in[2];
  const float* Wv = (const float*)d_in[3];
  const float* Wo = (const float*)d_in[4];
  const float* bo = (const float*)d_in[5];
  float* out = (float*)d_out;

  const size_t XE = (size_t)BDIM * NSEQ * DMODEL;  // 8,388,608
  const size_t WE = (size_t)DMODEL * DMODEL;       // 1,048,576

  bf16_t* xb   = (bf16_t*)d_ws;
  bf16_t* wqT  = xb  + XE;
  bf16_t* wkT  = wqT + WE;
  bf16_t* wvT  = wkT + WE;
  bf16_t* woT  = wvT + WE;
  bf16_t* qb   = woT + WE;
  bf16_t* kb   = qb  + XE;
  bf16_t* vtb  = kb  + XE;
  bf16_t* aob  = vtb + XE;  // total ~88 MB of workspace

  f32_to_bf16_kernel  <<<(int)((XE + 255) / 256), 256, 0, stream>>>(x,  xb, (int)XE);
  f32_to_bf16_T_kernel<<<(int)(WE / 256), 256, 0, stream>>>(Wq, wqT);
  f32_to_bf16_T_kernel<<<(int)(WE / 256), 256, 0, stream>>>(Wk, wkT);
  f32_to_bf16_T_kernel<<<(int)(WE / 256), 256, 0, stream>>>(Wv, wvT);
  f32_to_bf16_T_kernel<<<(int)(WE / 256), 256, 0, stream>>>(Wo, woT);

  dim3 gQKV(DMODEL / 64, (BDIM * NSEQ) / 128, 3);
  qkv_gemm_kernel<<<gQKV, 256, 0, stream>>>(xb, wqT, wkT, wvT, qb, kb, vtb);

  dim3 gAttn(NSEQ / 64, BDIM * NHEADS);
  attn_kernel<<<gAttn, 128, 0, stream>>>(qb, kb, vtb, aob);

  dim3 gOut(DMODEL / 64, (BDIM * NSEQ) / 128);
  out_gemm_kernel<<<gOut, 256, 0, stream>>>(aob, woT, bo, out);
}
